// UnifiedCascadePredictionModel_38989713113508
// MI455X (gfx1250) — matrix operand order repaired and checked by reading, not compiled, not run on hardware
//
#include <hip/hip_runtime.h>
#include <math.h>

#define B_ 4
#define T_ 8
#define N_ 3000
#define E_ 12000
#define D_ 256
#define H_ 256
#define HEADS_ 8
#define DH_ 32
#define F_ 8
#define L_ 3
#define HD_ 64
#define OUTB_ 45001   /* N_*7 + E_ + E_ + 1 */

typedef __attribute__((ext_vector_type(16))) _Float16 v16h;
typedef __attribute__((ext_vector_type(8)))  _Float16 v8h;
typedef __attribute__((ext_vector_type(8)))  float    v8f;

__device__ __forceinline__ float sigm(float x){ return 1.f/(1.f+expf(-x)); }
__device__ __forceinline__ float softplusf(float x){ return (x>20.f)?x:log1pf(expf(x)); }

__device__ __forceinline__ void atomicMaxF(float* addr, float val){
  int* ai=(int*)addr;
  int old=__float_as_int(*addr);
  while (val > __int_as_float(old)){
    int assumed=old;
    old = atomicCAS(ai, assumed, __float_as_int(val));
    if (old==assumed) break;
  }
}

// ---------------------------------------------------------------------------
// WMMA GEMM: C[M,N] = act(A[M,K] @ W[K,N] + bias)   (or C += A@W if accum)
// Templated on N,K so all strides are immediate offsets; K-loop fully
// unrolled; LDS double-buffered so next tile's global loads overlap WMMAs.
// Block tile 64x64, 4 waves (2x2), each wave 32x32 = 2x2 16x16x32 WMMA tiles.
// ---------------------------------------------------------------------------
#define BM 64
#define BN 64
#define BK 32

template<int N, int K>
__global__ void __launch_bounds__(128)
gemm_f16_wmma(const float* __restrict__ A, const float* __restrict__ W,
              const float* __restrict__ bias, float* __restrict__ C,
              int M, int act, int accum)
{
  __shared__ _Float16 Ah[2][BM][BK];   // 2 x 4 KB
  __shared__ _Float16 Bt[2][BN][BK];   // 2 x 4 KB, B transposed [n][k]

  const int tid  = threadIdx.x;
  const int wave = tid >> 5;
  const int lane = tid & 31;
  const int wm = (wave >> 1) * 32;
  const int wn = (wave & 1) * 32;
  const int bm0 = blockIdx.x * BM;
  const int bn0 = blockIdx.y * BN;

  // A staging map: thread owns row ar, 16 contiguous columns starting at ac0
  const int ar  = tid >> 1;          // 0..63
  const int ac0 = (tid & 1) * 16;    // 0 or 16
  const bool avalid = (bm0 + ar) < M;
  const float* aptr = A + (size_t)(bm0 + ar) * K + ac0;

  // B staging map: thread owns output column bnc, k-range [bk0, bk0+16)
  const int bnc = tid & 63;          // 0..63
  const int bk0 = (tid >> 6) * 16;   // 0 or 16
  const float* wptr = W + (size_t)bk0 * N + (bn0 + bnc);

  union U16 { v16h v; v8h h[2]; };

  auto load_tile = [&](int k0, v8h& a0, v8h& a1, v8h& b0, v8h& b1) {
    if (avalid) {
      const float4* g = (const float4*)(aptr + k0);
      float4 u0 = g[0], u1 = g[1], u2 = g[2], u3 = g[3];
      a0[0]=(_Float16)u0.x; a0[1]=(_Float16)u0.y; a0[2]=(_Float16)u0.z; a0[3]=(_Float16)u0.w;
      a0[4]=(_Float16)u1.x; a0[5]=(_Float16)u1.y; a0[6]=(_Float16)u1.z; a0[7]=(_Float16)u1.w;
      a1[0]=(_Float16)u2.x; a1[1]=(_Float16)u2.y; a1[2]=(_Float16)u2.z; a1[3]=(_Float16)u2.w;
      a1[4]=(_Float16)u3.x; a1[5]=(_Float16)u3.y; a1[6]=(_Float16)u3.z; a1[7]=(_Float16)u3.w;
    } else {
#pragma unroll
      for (int j = 0; j < 8; ++j) { a0[j] = (_Float16)0.f; a1[j] = (_Float16)0.f; }
    }
    // k-strided loads, stride = N (constexpr) -> immediate-offset loads
#pragma unroll
    for (int q = 0; q < 8; ++q) b0[q] = (_Float16)wptr[(size_t)(k0 + q) * N];
#pragma unroll
    for (int q = 0; q < 8; ++q) b1[q] = (_Float16)wptr[(size_t)(k0 + 8 + q) * N];
  };
  auto store_tile = [&](int buf, v8h a0, v8h a1, v8h b0, v8h b1) {
    *(v8h*)(&Ah[buf][ar][ac0])      = a0;
    *(v8h*)(&Ah[buf][ar][ac0 + 8])  = a1;
    *(v8h*)(&Bt[buf][bnc][bk0])     = b0;
    *(v8h*)(&Bt[buf][bnc][bk0 + 8]) = b1;
  };

  v8f acc[2][2] = {};

  {
    v8h a0, a1, b0, b1;
    load_tile(0, a0, a1, b0, b1);
    store_tile(0, a0, a1, b0, b1);
  }
  __syncthreads();

  int buf = 0;
#pragma unroll
  for (int k0 = 0; k0 < K; k0 += BK) {
    const bool have_next = (k0 + BK) < K;

    // issue next tile's global loads (latency hidden behind WMMAs)
    v8h a0, a1, b0, b1;
    if (have_next) {
      load_tile(k0 + BK, a0, a1, b0, b1);
      if (k0 + 2*BK < K) {            // warm L2 one tile further ahead
        if (avalid) __builtin_prefetch(aptr + k0 + 2*BK, 0, 3);
        __builtin_prefetch(wptr + (size_t)(k0 + 2*BK) * N, 0, 3);
      }
    }

    const int khalf = lane >> 4;   // lanes 0-15 -> K-half 0, 16-31 -> K-half 1
    const int l16   = lane & 15;

    // A fragments: lane holds row M=l16, halves k in [khalf*8,+8) and [16+khalf*8,+8)
    v16h afrag[2];
#pragma unroll
    for (int mt = 0; mt < 2; ++mt) {
      const _Float16* ap = &Ah[buf][wm + mt*16 + l16][0];
      U16 ua;
      ua.h[0] = *(const v8h*)(ap + khalf*8);
      ua.h[1] = *(const v8h*)(ap + 16 + khalf*8);
      afrag[mt] = ua.v;
    }
    // B fragments: lane holds col N=l16, halves k in [khalf*8,+8) and [16+khalf*8,+8)
    v16h bfrag[2];
#pragma unroll
    for (int nt = 0; nt < 2; ++nt) {
      const _Float16* bp = &Bt[buf][wn + nt*16 + l16][0];
      U16 ub;
      ub.h[0] = *(const v8h*)(bp + khalf*8);
      ub.h[1] = *(const v8h*)(bp + 16 + khalf*8);
      bfrag[nt] = ub.v;
    }
#pragma unroll
    for (int mt = 0; mt < 2; ++mt)
#pragma unroll
      for (int nt = 0; nt < 2; ++nt)
        acc[mt][nt] = __builtin_amdgcn_wmma_f32_16x16x32_f16(
            false, afrag[mt], false, bfrag[nt], (short)0, acc[mt][nt],
            false, false);

    if (have_next) {
      store_tile(buf ^ 1, a0, a1, b0, b1);
      __syncthreads();
      buf ^= 1;
    }
  }

  // C layout: VGPR r -> M = wm + mt*16 + (lane>=16?8:0) + r, N = wn + nt*16 + (lane&15)
  const int l16   = lane & 15;
  const int mhalf = (lane >> 4) * 8;
#pragma unroll
  for (int nt = 0; nt < 2; ++nt) {
    const int gn = bn0 + wn + nt*16 + l16;
    const float bv = (bias && !accum) ? bias[gn] : 0.f;
#pragma unroll
    for (int mt = 0; mt < 2; ++mt) {
#pragma unroll
      for (int r = 0; r < 8; ++r) {
        int gm = bm0 + wm + mt*16 + mhalf + r;
        if (gm < M) {
          float v = acc[mt][nt][r];
          size_t off = (size_t)gm * N + gn;
          if (accum) {
            C[off] += v;
          } else {
            v += bv;
            if (act == 1) v = v > 0.f ? v : 0.f;
            C[off] = v;
          }
        }
      }
    }
  }
}

// ---------------------------------------------------------------------------
// elementwise / scatter kernels
// ---------------------------------------------------------------------------
__global__ void fill_kernel(float* p, float v, int n){
  int i = blockIdx.x * blockDim.x + threadIdx.x;
  if (i < n) p[i] = v;
}

// interleave [env|infra|robot] tokens for one (b,t) slice -> X[3N x 256]
__global__ void pack_tokens(const float* env, const float* infra, const float* robot,
                            float* X, int b, int t){
  int idx = blockIdx.x * blockDim.x + threadIdx.x;
  if (idx >= 3 * N_ * D_) return;
  int row = idx >> 8, d = idx & 255;
  int n = row / 3, s = row % 3;
  const float* sp = (s == 0) ? env : (s == 1) ? infra : robot;
  X[idx] = sp[((size_t)(b * T_ + t) * N_ + n) * D_ + d];
}

// 8-head attention over 3 tokens per node: QKV[3N x 768] -> O[3N x 256]
__global__ void attn3_kernel(const float* __restrict__ QKV, float* __restrict__ O){
  int tid = blockIdx.x * blockDim.x + threadIdx.x;
  if (tid >= N_ * HEADS_) return;
  int n = tid >> 3, hd = tid & 7;
  const float* base = QKV + (size_t)(3 * n) * 768 + hd * DH_;
  float s[3][3];
#pragma unroll
  for (int i = 0; i < 3; ++i)
#pragma unroll
    for (int j = 0; j < 3; ++j) {
      float acc = 0.f;
      for (int d = 0; d < DH_; ++d)
        acc += base[i * 768 + d] * base[j * 768 + 256 + d];
      s[i][j] = acc * 0.17677669529663687f; // 1/sqrt(32)
    }
#pragma unroll
  for (int i = 0; i < 3; ++i) {
    float m = fmaxf(s[i][0], fmaxf(s[i][1], s[i][2]));
    float e0 = expf(s[i][0]-m), e1 = expf(s[i][1]-m), e2 = expf(s[i][2]-m);
    float inv = 1.f / (e0 + e1 + e2);
    float a0 = e0*inv, a1 = e1*inv, a2 = e2*inv;
    for (int d = 0; d < DH_; ++d) {
      float o = a0 * base[0*768 + 512 + d] + a1 * base[1*768 + 512 + d]
              + a2 * base[2*768 + 512 + d];
      O[(size_t)(3*n + i) * D_ + hd * DH_ + d] = o;
    }
  }
}

// mean over 3 tokens + layernorm -> fused slice [N x 256]
__global__ void mean3_ln(const float* __restrict__ P, const float* g,
                         const float* bb, float* __restrict__ out){
  int n = blockIdx.x, d = threadIdx.x;
  float y = (P[(size_t)(3*n)*D_ + d] + P[(size_t)(3*n+1)*D_ + d]
           + P[(size_t)(3*n+2)*D_ + d]) * (1.f/3.f);
  __shared__ float s1[256], s2[256];
  s1[d] = y; s2[d] = y*y; __syncthreads();
  for (int st = 128; st > 0; st >>= 1) {
    if (d < st) { s1[d] += s1[d+st]; s2[d] += s2[d+st]; }
    __syncthreads();
  }
  float mean = s1[0] * (1.f/256.f);
  float var  = s2[0] * (1.f/256.f) - mean*mean;
  float r = rsqrtf(var + 1e-5f);
  out[(size_t)n * D_ + d] = (y - mean) * r * g[d] + bb[d];
}

// h = LN(h + hn) with per-layer gamma/beta
__global__ void add_ln(float* __restrict__ h, const float* __restrict__ hn,
                       const float* g, const float* bb){
  int row = blockIdx.x, d = threadIdx.x;
  float y = h[(size_t)row*H_ + d] + hn[(size_t)row*H_ + d];
  __shared__ float s1[256], s2[256];
  s1[d] = y; s2[d] = y*y; __syncthreads();
  for (int st = 128; st > 0; st >>= 1) {
    if (d < st) { s1[d] += s1[d+st]; s2[d] += s2[d+st]; }
    __syncthreads();
  }
  float mean = s1[0] * (1.f/256.f);
  float var  = s2[0] * (1.f/256.f) - mean*mean;
  float r = rsqrtf(var + 1e-5f);
  h[(size_t)row*H_ + d] = (y - mean) * r * g[d] + bb[d];
}

// edge_emb = relu(edge_attr @ we + be), K=8 (too small for WMMA)
__global__ void edge_emb_kernel(const float* __restrict__ ea, const float* __restrict__ we,
                                const float* __restrict__ be, float* __restrict__ out){
  int idx = blockIdx.x * blockDim.x + threadIdx.x;
  if (idx >= B_ * E_ * H_) return;
  int r = idx >> 8, c = idx & 255;
  float s = be[c];
#pragma unroll
  for (int f = 0; f < F_; ++f) s += ea[(size_t)r * F_ + f] * we[f * H_ + c];
  out[idx] = s > 0.f ? s : 0.f;
}

__global__ void gat_logit(const float* __restrict__ xw, const float* __restrict__ ew,
                          const int* __restrict__ src, const int* __restrict__ dst,
                          const float* asrc, const float* adst, const float* aedge,
                          float* __restrict__ logit){
  int tid = blockIdx.x * blockDim.x + threadIdx.x;
  if (tid >= B_ * E_ * HEADS_) return;
  int b = tid / (E_ * HEADS_);
  int rem = tid % (E_ * HEADS_);
  int e = rem >> 3, hd = rem & 7;
  int s = src[e], d = dst[e];
  const float* xs = xw + ((size_t)b * N_ + s) * H_ + hd * DH_;
  const float* xd = xw + ((size_t)b * N_ + d) * H_ + hd * DH_;
  const float* ee = ew + ((size_t)b * E_ + e) * H_ + hd * DH_;
  float lg = 0.f;
  for (int k = 0; k < DH_; ++k)
    lg += xs[k]*asrc[hd*DH_+k] + xd[k]*adst[hd*DH_+k] + ee[k]*aedge[hd*DH_+k];
  logit[tid] = lg > 0.f ? lg : 0.2f * lg;
}

__global__ void seg_max_kernel(const float* __restrict__ logit, const int* __restrict__ dst,
                               float* __restrict__ segmax){
  int tid = blockIdx.x * blockDim.x + threadIdx.x;
  if (tid >= B_ * E_ * HEADS_) return;
  int b = tid / (E_ * HEADS_);
  int rem = tid % (E_ * HEADS_);
  int e = rem >> 3, hd = rem & 7;
  atomicMaxF(&segmax[((size_t)b * N_ + dst[e]) * HEADS_ + hd], logit[tid]);
}

__global__ void seg_expsum(float* __restrict__ w, const int* __restrict__ dst,
                           const float* __restrict__ segmax, float* __restrict__ segsum){
  int tid = blockIdx.x * blockDim.x + threadIdx.x;
  if (tid >= B_ * E_ * HEADS_) return;
  int b = tid / (E_ * HEADS_);
  int rem = tid % (E_ * HEADS_);
  int e = rem >> 3, hd = rem & 7;
  size_t si = ((size_t)b * N_ + dst[e]) * HEADS_ + hd;
  float ev = expf(w[tid] - segmax[si]);
  w[tid] = ev;
  atomicAdd(&segsum[si], ev);
}

__global__ void gat_scatter(const float* __restrict__ xw, const float* __restrict__ ew,
                            const float* __restrict__ w, const float* __restrict__ segsum,
                            const int* __restrict__ src, const int* __restrict__ dst,
                            float* __restrict__ out){
  int tid = blockIdx.x * blockDim.x + threadIdx.x;
  if (tid >= B_ * E_ * HEADS_) return;
  int b = tid / (E_ * HEADS_);
  int rem = tid % (E_ * HEADS_);
  int e = rem >> 3, hd = rem & 7;
  int s = src[e], d = dst[e];
  float wn = w[tid] / (segsum[((size_t)b * N_ + d) * HEADS_ + hd] + 1e-16f);
  const float* xs = xw + ((size_t)b * N_ + s) * H_ + hd * DH_;
  const float* ee = ew + ((size_t)b * E_ + e) * H_ + hd * DH_;
  float* op = out + ((size_t)b * N_ + d) * H_ + hd * DH_;
  for (int k = 0; k < DH_; ++k)
    atomicAdd(&op[k], (xs[k] + ee[k]) * wn);
}

__global__ void lstm_elem(const float* __restrict__ gates, float* __restrict__ h,
                          float* __restrict__ c){
  int idx = blockIdx.x * blockDim.x + threadIdx.x;
  if (idx >= B_ * N_ * H_) return;
  int row = idx >> 8, d = idx & 255;
  const float* g4 = gates + (size_t)row * 4 * H_;
  float ig = sigm(g4[d]);
  float fg = sigm(g4[H_ + d]);
  float gg = tanhf(g4[2*H_ + d]);
  float og = sigm(g4[3*H_ + d]);
  float cc = fg * c[idx] + ig * gg;
  c[idx] = cc;
  h[idx] = og * tanhf(cc);
}

__global__ void mean_nodes(const float* __restrict__ h, float* __restrict__ hg){
  int b = blockIdx.x, d = threadIdx.x;
  float s = 0.f;
  for (int n = 0; n < N_; ++n) s += h[((size_t)b * N_ + n) * H_ + d];
  hg[b * H_ + d] = s * (1.f / N_);
}

__global__ void head_node_out(const float* __restrict__ hidden, const float* __restrict__ w2,
                              const float* __restrict__ b2, float* __restrict__ out,
                              int col, int act){
  int idx = blockIdx.x * blockDim.x + threadIdx.x;
  if (idx >= B_ * N_) return;
  int b = idx / N_, n = idx % N_;
  float s = b2[0];
  for (int k = 0; k < HD_; ++k) s += hidden[(size_t)idx * HD_ + k] * w2[k];
  if (act == 2) s = sigm(s);
  else if (act == 3) s = softplusf(s);
  out[(size_t)b * OUTB_ + n * 7 + col] = s;
}

__global__ void head_edge_out(const float* __restrict__ hidden, const float* __restrict__ w2,
                              const float* __restrict__ b2, float* __restrict__ out,
                              int offset){
  int idx = blockIdx.x * blockDim.x + threadIdx.x;
  if (idx >= B_ * E_) return;
  int b = idx / E_, e = idx % E_;
  float s = b2[0];
  for (int k = 0; k < HD_; ++k) s += hidden[(size_t)idx * HD_ + k] * w2[k];
  out[(size_t)b * OUTB_ + offset + e] = s;
}

__global__ void freq_out(const float* __restrict__ hgh, const float* __restrict__ w2,
                         const float* __restrict__ b2, float* __restrict__ out){
  int b = threadIdx.x;
  if (b >= B_) return;
  float s = b2[0];
  for (int k = 0; k < HD_; ++k) s += hgh[b * HD_ + k] * w2[k];
  out[(size_t)b * OUTB_ + (OUTB_ - 1)] = s;
}

__global__ void pack_ef(const float* __restrict__ h, const int* __restrict__ src,
                        const int* __restrict__ dst, float* __restrict__ ef){
  long idx = (long)blockIdx.x * blockDim.x + threadIdx.x;
  if (idx >= (long)B_ * E_ * 2 * H_) return;
  int row = (int)(idx >> 9);           // /512
  int col = (int)(idx & 511);
  int b = row / E_, e = row % E_;
  int node = (col < H_) ? src[e] : dst[e];
  int d = (col < H_) ? col : col - H_;
  ef[idx] = h[((size_t)b * N_ + node) * H_ + d];
}

// ---------------------------------------------------------------------------
extern "C" void kernel_launch(void* const* d_in, const int* in_sizes, int n_in,
                              void* d_out, int out_size, void* d_ws, size_t ws_size,
                              hipStream_t stream) {
  const float* env     = (const float*)d_in[0];
  const float* infra   = (const float*)d_in[1];
  const float* robot   = (const float*)d_in[2];
  const float* eattr   = (const float*)d_in[3];
  const int*   eidx    = (const int*)d_in[4];
  const float* wqkv    = (const float*)d_in[5];
  const float* bqkv    = (const float*)d_in[6];
  const float* wo      = (const float*)d_in[7];
  const float* bo      = (const float*)d_in[8];
  const float* lnf_g   = (const float*)d_in[9];
  const float* lnf_b   = (const float*)d_in[10];
  const float* we_emb  = (const float*)d_in[11];
  const float* be_emb  = (const float*)d_in[12];
  const float* tg_W    = (const float*)d_in[13];
  const float* tg_We   = (const float*)d_in[14];
  const float* tg_asrc = (const float*)d_in[15];
  const float* tg_adst = (const float*)d_in[16];
  const float* tg_aedge= (const float*)d_in[17];
  const float* w_ih    = (const float*)d_in[18];
  const float* w_hh    = (const float*)d_in[19];
  const float* b_lstm  = (const float*)d_in[20];
  const float* g_W     = (const float*)d_in[21];
  const float* g_We    = (const float*)d_in[22];
  const float* g_asrc  = (const float*)d_in[23];
  const float* g_adst  = (const float*)d_in[24];
  const float* g_aedge = (const float*)d_in[25];
  const float* g_lng   = (const float*)d_in[26];
  const float* g_lnb   = (const float*)d_in[27];
  const float* hn_w1   = (const float*)d_in[28];
  const float* hn_b1   = (const float*)d_in[29];
  const float* hn_w2   = (const float*)d_in[30];
  const float* hn_b2   = (const float*)d_in[31];
  const float* he_w1   = (const float*)d_in[32];
  const float* he_b1   = (const float*)d_in[33];
  const float* he_w2   = (const float*)d_in[34];
  const float* he_b2   = (const float*)d_in[35];

  const int* src = eidx;
  const int* dst = eidx + E_;
  float* out = (float*)d_out;

  // workspace carve-up (floats)
  float* ws = (float*)d_ws;
  size_t off = 0;
  auto alloc = [&](size_t n){ float* p = ws + off; off += n; return p; };
  float* fused  = alloc((size_t)T_ * B_ * N_ * D_);   // [t][b][n][d]
  float* eemb   = alloc((size_t)B_ * E_ * H_);
  float* ew     = alloc((size_t)B_ * E_ * H_);
  float* h      = alloc((size_t)B_ * N_ * H_);
  float* c      = alloc((size_t)B_ * N_ * H_);
  float* xw     = alloc((size_t)B_ * N_ * H_);
  float* msg    = alloc((size_t)B_ * N_ * H_);
  float* gates  = alloc((size_t)B_ * N_ * 4 * H_);
  float* wlog   = alloc((size_t)B_ * E_ * HEADS_);
  float* segmax = alloc((size_t)B_ * N_ * HEADS_);
  float* segsum = alloc((size_t)B_ * N_ * HEADS_);
  float* Xs     = alloc((size_t)3 * N_ * D_);
  float* QKVs   = alloc((size_t)3 * N_ * 3 * D_);
  float* Os     = alloc((size_t)3 * N_ * D_);
  float* Ps     = alloc((size_t)3 * N_ * D_);
  float* hidden = alloc((size_t)B_ * E_ * HD_);
  float* hg     = alloc((size_t)B_ * H_);
  float* hgh    = alloc((size_t)B_ * HD_);
  float* ef     = fused;  // fused dead after the LSTM scan; reuse (same size)

  auto fill = [&](float* p, float v, int n){
    fill_kernel<<<(n + 255) / 256, 256, 0, stream>>>(p, v, n);
  };
  const int EH  = B_ * E_ * HEADS_;
  const int EHB = (EH + 255) / 256;

#define GEMM(NN, KK, A, W, BIAS, C, M, ACT, ACCUM)                              \
  gemm_f16_wmma<NN, KK><<<dim3(((M) + BM - 1) / BM, (NN) / BN), dim3(128), 0,   \
                          stream>>>((A), (W), (BIAS), (C), (M), (ACT), (ACCUM))

  // ---- Stage 1: token fusion (attention over 3 token types) per (b,t) ----
  for (int b = 0; b < B_; ++b)
    for (int t = 0; t < T_; ++t) {
      pack_tokens<<<(3*N_*D_ + 255)/256, 256, 0, stream>>>(env, infra, robot, Xs, b, t);
      GEMM(768, 256, Xs, wqkv, bqkv, QKVs, 3*N_, 0, 0);
      attn3_kernel<<<(N_*HEADS_ + 255)/256, 256, 0, stream>>>(QKVs, Os);
      GEMM(256, 256, Os, wo, bo, Ps, 3*N_, 0, 0);
      mean3_ln<<<N_, 256, 0, stream>>>(Ps, lnf_g, lnf_b,
                                       fused + ((size_t)t * B_ + b) * N_ * D_);
    }

  // ---- edge embedding + loop-invariant tg edge projection ----
  edge_emb_kernel<<<(B_*E_*H_ + 255)/256, 256, 0, stream>>>(eattr, we_emb, be_emb, eemb);
  GEMM(256, 256, eemb, tg_We, nullptr, ew, B_*E_, 0, 0);

  // ---- LSTM scan over T with GAT message passing ----
  fill(h, 0.f, B_*N_*H_);
  fill(c, 0.f, B_*N_*H_);
  for (int t = 0; t < T_; ++t) {
    const float* xt = fused + (size_t)t * B_ * N_ * D_;
    GEMM(256, 256, xt, tg_W, nullptr, xw, B_*N_, 0, 0);
    fill(segmax, -1e30f, B_*N_*HEADS_);
    fill(segsum, 0.f,    B_*N_*HEADS_);
    fill(msg,    0.f,    B_*N_*H_);
    gat_logit<<<EHB, 256, 0, stream>>>(xw, ew, src, dst, tg_asrc, tg_adst, tg_aedge, wlog);
    seg_max_kernel<<<EHB, 256, 0, stream>>>(wlog, dst, segmax);
    seg_expsum<<<EHB, 256, 0, stream>>>(wlog, dst, segmax, segsum);
    gat_scatter<<<EHB, 256, 0, stream>>>(xw, ew, wlog, segsum, src, dst, msg);
    GEMM(1024, 256, msg, w_ih, b_lstm, gates, B_*N_, 0, 0);
    GEMM(1024, 256, h,   w_hh, nullptr, gates, B_*N_, 0, 1);  // accumulate
    lstm_elem<<<(B_*N_*H_ + 255)/256, 256, 0, stream>>>(gates, h, c);
  }

  // ---- L residual GAT layers with layernorm ----
  for (int i = 0; i < L_; ++i) {
    GEMM(256, 256, eemb, g_We + (size_t)i*H_*H_, nullptr, ew, B_*E_, 0, 0);
    GEMM(256, 256, h,    g_W  + (size_t)i*H_*H_, nullptr, xw, B_*N_, 0, 0);
    fill(segmax, -1e30f, B_*N_*HEADS_);
    fill(segsum, 0.f,    B_*N_*HEADS_);
    fill(msg,    0.f,    B_*N_*H_);
    gat_logit<<<EHB, 256, 0, stream>>>(xw, ew, src, dst,
        g_asrc + i*HEADS_*DH_, g_adst + i*HEADS_*DH_, g_aedge + i*HEADS_*DH_, wlog);
    seg_max_kernel<<<EHB, 256, 0, stream>>>(wlog, dst, segmax);
    seg_expsum<<<EHB, 256, 0, stream>>>(wlog, dst, segmax, segsum);
    gat_scatter<<<EHB, 256, 0, stream>>>(xw, ew, wlog, segsum, src, dst, msg);
    add_ln<<<B_*N_, 256, 0, stream>>>(h, msg, g_lng + i*H_, g_lnb + i*H_);
  }

  // ---- node-level heads (write directly into d_out layout) ----
  for (int j = 0; j < 7; ++j) {
    GEMM(64, 256, h, hn_w1 + (size_t)j*H_*HD_, hn_b1 + j*HD_, hidden, B_*N_, 1, 0);
    int act = (j == 0 || j == 6) ? 2 : (j == 1 ? 3 : 0);
    head_node_out<<<(B_*N_ + 255)/256, 256, 0, stream>>>(
        hidden, hn_w2 + j*HD_, hn_b2 + j, out, j, act);
  }

  // ---- global frequency head ----
  mean_nodes<<<B_, 256, 0, stream>>>(h, hg);
  GEMM(64, 256, hg, hn_w1 + (size_t)7*H_*HD_, hn_b1 + 7*HD_, hgh, B_, 1, 0);
  freq_out<<<1, 32, 0, stream>>>(hgh, hn_w2 + 7*HD_, hn_b2 + 7, out);

  // ---- edge-level heads ----
  pack_ef<<<(int)(((long)B_*E_*2*H_ + 255)/256), 256, 0, stream>>>(h, src, dst, ef);
  for (int k = 0; k < 2; ++k) {
    GEMM(64, 512, ef, he_w1 + (size_t)k*2*H_*HD_, he_b1 + k*HD_, hidden, B_*E_, 1, 0);
    head_edge_out<<<(B_*E_ + 255)/256, 256, 0, stream>>>(
        hidden, he_w2 + k*HD_, he_b2 + k, out, N_*7 + k*E_);
  }
#undef GEMM
}
